// Model2_52656299049198
// MI455X (gfx1250) — matrix operand order repaired and checked by reading, not compiled, not run on hardware
//
#include <hip/hip_runtime.h>
#include <hip/hip_bf16.h>

typedef __attribute__((ext_vector_type(16))) _Float16 v16h;
typedef __attribute__((ext_vector_type(8)))  float    v8f;

#define H   100
#define T   256
#define V   100000
#define NL  1000
#define S   4096
#define MAXLEN 50
#define E2  200
#define G3  300          // 3*H
#define KP_K 224         // padded K for attention GEMMs (7 chunks of 32)

// ---------------------------------------------------------------------------
// f32 -> f16 matrix convert with optional zero-padded row stride
// ---------------------------------------------------------------------------
__global__ void convert_f16_kernel(const float* __restrict__ src, _Float16* __restrict__ dst,
                                   int rows, int cols, int dstStride) {
    int idx = blockIdx.x * blockDim.x + threadIdx.x;
    int total = rows * dstStride;
    if (idx >= total) return;
    int r = idx / dstStride, c = idx % dstStride;
    float v = (c < cols) ? src[r * cols + c] : 0.0f;
    dst[idx] = (_Float16)v;
}

// ---------------------------------------------------------------------------
// Embedding gather -> f16 activation matrix (S x T)
// ---------------------------------------------------------------------------
__global__ void embed_f16_kernel(const int* __restrict__ x, const float* __restrict__ E,
                                 _Float16* __restrict__ emb16) {
    int t = blockIdx.x;          // 0..S-1
    int c = threadIdx.x;         // 0..T-1
    emb16[(long)t * T + c] = (_Float16)E[(long)x[t] * T + c];
}

// zero the pad columns [E2, KP_K) of the f16 encoded matrix
__global__ void pad_enc16_kernel(_Float16* __restrict__ enc16) {
    int t = blockIdx.x;
    int c = threadIdx.x;         // 0..31
    if (c < KP_K - E2) enc16[(long)t * KP_K + E2 + c] = (_Float16)0.0f;
}

// ---------------------------------------------------------------------------
// WMMA GEMM:  C[m][n] = bias[n] + sum_k A[m'][k] * B[n][k]
//   A: f16, row stride lda, m' = flipA ? (Arows-1-m) : m
//   B: f16 weight matrix stored row-major (n, k), row stride ldb
//   C: f32, row stride ldc
// Each wave computes a 16(M) x 32(N) tile: 1 A fragment, 2 B fragments,
// 2 accumulators -> 2 independent v_wmma per k-chunk.
// grid: (ceil(N/32), M/64), block: 128 threads (4 waves, 1 m-tile each).
// All loads unconditional (indices clamped); EXEC stays all-ones.
// ---------------------------------------------------------------------------
__global__ void wmma_gemm_kernel(const _Float16* __restrict__ A, int lda, int Arows, int flipA,
                                 const _Float16* __restrict__ B, int ldb,
                                 const float* __restrict__ bias,
                                 float* __restrict__ C, int ldc,
                                 int N, int kChunks) {
    int wave = threadIdx.x >> 5;           // 0..3
    int lane = threadIdx.x & 31;
    int lm = lane & 15;                    // 0..15
    int lh = lane >> 4;                    // 0..1
    int m0 = (blockIdx.y * 4 + wave) * 16;
    int n0 = blockIdx.x * 32;

    int arow = m0 + lm;
    if (flipA) arow = Arows - 1 - arow;

    // B row indices for the two column tiles; clamp so every load is valid.
    // (Out-of-range columns produce garbage that the D-store mask discards.)
    int bn0 = n0 + lm;       if (bn0 > N - 1) bn0 = N - 1;
    int bn1 = n0 + 16 + lm;  if (bn1 > N - 1) bn1 = N - 1;

    // A fragment halves: r=0..3 contiguous at k0+8h, r=4..7 at k0+16+8h.
    // B fragment halves: r=0..3 contiguous at k0+8h, r=4..7 at k0+16+8h.
    const _Float16* pa  = A + (long)arow * lda + 8 * lh;
    const _Float16* pb0 = B + (long)bn0 * ldb + 8 * lh;
    const _Float16* pb1 = B + (long)bn1 * ldb + 8 * lh;

    v8f acc0 = {};
    v8f acc1 = {};
    for (int kc = 0; kc < kChunks; ++kc) {
        union { v16h v; uint4 q[2]; } a, b0, b1;
        a.q[0]  = *(const uint4*)(pa);
        a.q[1]  = *(const uint4*)(pa + 16);
        b0.q[0] = *(const uint4*)(pb0);
        b0.q[1] = *(const uint4*)(pb0 + 16);
        b1.q[0] = *(const uint4*)(pb1);
        b1.q[1] = *(const uint4*)(pb1 + 16);
        acc0 = __builtin_amdgcn_wmma_f32_16x16x32_f16(
            false, a.v, false, b0.v, (short)0, acc0, false, false);
        acc1 = __builtin_amdgcn_wmma_f32_16x16x32_f16(
            false, a.v, false, b1.v, (short)0, acc1, false, false);
        pa += 32; pb0 += 32; pb1 += 32;
    }

    // D (16x16 f32): lane L col = L%16, VGPR r row = r + 8h
    int cn0 = n0 + lm;
    int cn1 = n0 + 16 + lm;
    if (cn0 < N) {
        float bv = bias[cn0];
#pragma unroll
        for (int r = 0; r < 8; ++r)
            C[(long)(m0 + r + 8 * lh) * ldc + cn0] = acc0[r] + bv;
    }
    if (cn1 < N) {
        float bv = bias[cn1];
#pragma unroll
        for (int r = 0; r < 8; ++r)
            C[(long)(m0 + r + 8 * lh) * ldc + cn1] = acc1[r] + bv;
    }
}

// ---------------------------------------------------------------------------
// Sequential GRU scans: one persistent block per direction, h in LDS.
// gi (S x 300) precomputed (includes bih). Writes encoded f32 (S x 200)
// and f16 (S x KP_K, cols <200).
// block: 320 threads
// ---------------------------------------------------------------------------
__device__ __forceinline__ float sigm(float x) { return 1.0f / (1.0f + expf(-x)); }

__global__ void gru_recurrent_kernel(const float* __restrict__ Whh_f, const float* __restrict__ bhh_f,
                                     const float* __restrict__ Whh_b, const float* __restrict__ bhh_b,
                                     const float* __restrict__ gi_f, const float* __restrict__ gi_b,
                                     float* __restrict__ enc, _Float16* __restrict__ enc16) {
    int dir = blockIdx.x;                         // 0 = forward, 1 = backward
    const float* Whh = dir ? Whh_b : Whh_f;
    const float* bhh = dir ? bhh_b : bhh_f;
    const float* gi  = dir ? gi_b  : gi_f;

    __shared__ float h[H];
    __shared__ float gh[G3];
    int tid = threadIdx.x;
    if (tid < H) h[tid] = 0.0f;
    __syncthreads();

    for (int s = 0; s < S; ++s) {
        if (tid < G3) {
            float acc = bhh[tid];
            const float* wr = Whh + tid * H;
            for (int k = 0; k < H; ++k) acc += wr[k] * h[k];
            gh[tid] = acc;
        }
        __syncthreads();
        float hn = 0.0f;
        if (tid < H) {
            const float* g = gi + (long)s * G3;
            float r = sigm(g[tid]         + gh[tid]);
            float z = sigm(g[H + tid]     + gh[H + tid]);
            float n = tanhf(g[2 * H + tid] + r * gh[2 * H + tid]);
            hn = (1.0f - z) * n + z * h[tid];
        }
        __syncthreads();
        if (tid < H) {
            h[tid] = hn;
            int t = dir ? (S - 1 - s) : s;          // backward outputs are reversed
            enc[(long)t * E2 + dir * H + tid] = hn;
            enc16[(long)t * KP_K + dir * H + tid] = (_Float16)hn;
        }
        __syncthreads();
    }
}

// ---------------------------------------------------------------------------
// Decoder: single block, 1024 threads, 50 sequential steps.
// Note: reference calls gru_cell with h=0, so gh == bhh and h' = (1-z)*n.
// ---------------------------------------------------------------------------
__global__ void __launch_bounds__(1024)
decoder_kernel(const float* __restrict__ enc,
               const float* __restrict__ kp, const float* __restrict__ vp,
               const float* __restrict__ mid_W, const float* __restrict__ mid_b,
               const float* __restrict__ att_in_W, const float* __restrict__ att_in_b,
               const float* __restrict__ att_out_W, const float* __restrict__ att_out_b,
               const float* __restrict__ dec_Wih, const float* __restrict__ dec_bih,
               const float* __restrict__ dec_bhh,
               const float* __restrict__ last_W, const float* __restrict__ last_b,
               const float* __restrict__ L_tab, float* __restrict__ out) {
    __shared__ float att[S];
    __shared__ float red[1024];
    __shared__ int   redi[1024];
    __shared__ float hfull[E2], q[E2], ctx[E2], attv[E2], gi[G3], hd[H];
    __shared__ float pctx[4][E2];
    int tid = threadIdx.x;

    // sent = mid_W @ concat(encoded[0, H:], encoded[-1, :H]) + mid_b ; h0 = [0, sent]
    if (tid < H) {
        float acc = mid_b[tid];
        const float* wr = mid_W + tid * E2;
        for (int k = 0; k < H; ++k) acc += wr[k]       * enc[(long)0 * E2 + H + k];
        for (int k = 0; k < H; ++k) acc += wr[H + k]   * enc[(long)(S - 1) * E2 + k];
        hfull[H + tid] = acc;
        hfull[tid] = 0.0f;
    }
    __syncthreads();

    const float scale = 0.07071067811865475f;   // 1/sqrt(200)
    const float* Wq = att_in_W;                 // rows [0,200)
    const float* bq = att_in_b;

    for (int step = 0; step < MAXLEN; ++step) {
        // q = Wq @ h + bq
        if (tid < E2) {
            float acc = bq[tid];
            const float* wr = Wq + tid * E2;
            for (int k = 0; k < E2; ++k) acc += wr[k] * hfull[k];
            q[tid] = acc;
        }
        __syncthreads();
        // scores = kp @ q * scale ; track max
        float lmax = -3.0e38f;
        for (int t = tid; t < S; t += 1024) {
            const float* kr = kp + (long)t * E2;
            float acc = 0.0f;
            for (int k = 0; k < E2; ++k) acc += kr[k] * q[k];
            acc *= scale;
            att[t] = acc;
            lmax = fmaxf(lmax, acc);
        }
        red[tid] = lmax;
        __syncthreads();
        for (int s2 = 512; s2 > 0; s2 >>= 1) {
            if (tid < s2) red[tid] = fmaxf(red[tid], red[tid + s2]);
            __syncthreads();
        }
        float mx = red[0];
        __syncthreads();
        float lsum = 0.0f;
        for (int t = tid; t < S; t += 1024) {
            float e = expf(att[t] - mx);
            att[t] = e;
            lsum += e;
        }
        red[tid] = lsum;
        __syncthreads();
        for (int s2 = 512; s2 > 0; s2 >>= 1) {
            if (tid < s2) red[tid] += red[tid + s2];
            __syncthreads();
        }
        float inv = 1.0f / red[0];
        __syncthreads();
        // ctx = softmax(att) @ vp   (4 t-partitions in parallel)
        {
            int c = tid & 255, g = tid >> 8;
            if (c < E2) {
                float acc = 0.0f;
                int t0 = g * (S / 4);
                for (int t = t0; t < t0 + S / 4; ++t) acc += att[t] * vp[(long)t * E2 + c];
                pctx[g][c] = acc;
            }
        }
        __syncthreads();
        if (tid < E2) ctx[tid] = (pctx[0][tid] + pctx[1][tid] + pctx[2][tid] + pctx[3][tid]) * inv;
        __syncthreads();
        // att = att_out_W @ ctx + att_out_b
        if (tid < E2) {
            float acc = att_out_b[tid];
            const float* wr = att_out_W + tid * E2;
            for (int k = 0; k < E2; ++k) acc += wr[k] * ctx[k];
            attv[tid] = acc;
        }
        __syncthreads();
        // gi = dec_Wih @ concat(h, att) + dec_bih
        if (tid < G3) {
            float acc = dec_bih[tid];
            const float* wr = dec_Wih + tid * (2 * E2);
            for (int k = 0; k < E2; ++k) acc += wr[k]      * hfull[k];
            for (int k = 0; k < E2; ++k) acc += wr[E2 + k] * attv[k];
            gi[tid] = acc;
        }
        __syncthreads();
        // GRU cell with h=0: gh == bhh, h' = (1-z)*n
        if (tid < H) {
            float r = sigm(gi[tid]         + dec_bhh[tid]);
            float z = sigm(gi[H + tid]     + dec_bhh[H + tid]);
            float n = tanhf(gi[2 * H + tid] + r * dec_bhh[2 * H + tid]);
            hd[tid] = (1.0f - z) * n;
        }
        __syncthreads();
        // scores = last_W @ hd + last_b ; write out ; argmax (first max wins)
        float sv = -3.0e38f; int si = 0;
        if (tid < NL) {
            float acc = last_b[tid];
            const float* wr = last_W + tid * H;
            for (int k = 0; k < H; ++k) acc += wr[k] * hd[k];
            out[step * NL + tid] = acc;
            sv = acc; si = tid;
        }
        red[tid] = sv; redi[tid] = si;
        __syncthreads();
        for (int s2 = 512; s2 > 0; s2 >>= 1) {
            if (tid < s2) {
                float ov = red[tid + s2]; int oi = redi[tid + s2];
                if (ov > red[tid] || (ov == red[tid] && oi < redi[tid])) {
                    red[tid] = ov; redi[tid] = oi;
                }
            }
            __syncthreads();
        }
        int idx = redi[0];
        __syncthreads();
        if (tid < H) {
            hfull[tid]     = hd[tid];
            hfull[H + tid] = L_tab[(long)idx * H + tid];
        }
        __syncthreads();
    }
}

// ---------------------------------------------------------------------------
// host-side launch
// ---------------------------------------------------------------------------
extern "C" void kernel_launch(void* const* d_in, const int* in_sizes, int n_in,
                              void* d_out, int out_size, void* d_ws, size_t ws_size,
                              hipStream_t stream) {
    const int*   x         = (const int*)  d_in[0];
    const float* E_tab     = (const float*)d_in[1];
    const float* L_tab     = (const float*)d_in[2];
    const float* enc_f_Wih = (const float*)d_in[3];
    const float* enc_f_Whh = (const float*)d_in[4];
    const float* enc_f_bih = (const float*)d_in[5];
    const float* enc_f_bhh = (const float*)d_in[6];
    const float* enc_b_Wih = (const float*)d_in[7];
    const float* enc_b_Whh = (const float*)d_in[8];
    const float* enc_b_bih = (const float*)d_in[9];
    const float* enc_b_bhh = (const float*)d_in[10];
    const float* mid_W     = (const float*)d_in[11];
    const float* mid_b     = (const float*)d_in[12];
    // d_in[13], d_in[14]: len_W / len_b — dead code in the reference (lscores unused)
    const float* dec_Wih   = (const float*)d_in[15];
    // d_in[16]: dec_Whh — multiplied by h=0, only bhh survives
    const float* dec_bih   = (const float*)d_in[17];
    const float* dec_bhh   = (const float*)d_in[18];
    const float* last_W    = (const float*)d_in[19];
    const float* last_b    = (const float*)d_in[20];
    const float* att_in_W  = (const float*)d_in[21];
    const float* att_in_b  = (const float*)d_in[22];
    const float* att_out_W = (const float*)d_in[23];
    const float* att_out_b = (const float*)d_in[24];
    float* out = (float*)d_out;

    // ---- workspace carve-up (256B aligned) ----
    char* p = (char*)d_ws;
    auto take = [&](size_t bytes) -> char* {
        char* r = p;
        p += (bytes + 255) & ~size_t(255);
        return r;
    };
    _Float16* emb16   = (_Float16*)take((size_t)S * T * 2);
    _Float16* wihf16  = (_Float16*)take((size_t)G3 * T * 2);
    _Float16* wihb16  = (_Float16*)take((size_t)G3 * T * 2);
    _Float16* wk16    = (_Float16*)take((size_t)E2 * KP_K * 2);
    _Float16* wv16    = (_Float16*)take((size_t)E2 * KP_K * 2);
    float*    gi_f    = (float*)   take((size_t)S * G3 * 4);
    float*    gi_b    = (float*)   take((size_t)S * G3 * 4);
    float*    enc     = (float*)   take((size_t)S * E2 * 4);
    _Float16* enc16   = (_Float16*)take((size_t)S * KP_K * 2);
    float*    kp      = (float*)   take((size_t)S * E2 * 4);
    float*    vp      = (float*)   take((size_t)S * E2 * 4);
    (void)ws_size; (void)n_in; (void)in_sizes; (void)out_size;

    // 1) weight conversions (f32 -> f16, K-padded for attention weights)
    {
        int tot = G3 * T;
        convert_f16_kernel<<<(tot + 255) / 256, 256, 0, stream>>>(enc_f_Wih, wihf16, G3, T, T);
        convert_f16_kernel<<<(tot + 255) / 256, 256, 0, stream>>>(enc_b_Wih, wihb16, G3, T, T);
        int tota = E2 * KP_K;
        convert_f16_kernel<<<(tota + 255) / 256, 256, 0, stream>>>(att_in_W + (size_t)E2 * E2,     wk16, E2, E2, KP_K);
        convert_f16_kernel<<<(tota + 255) / 256, 256, 0, stream>>>(att_in_W + (size_t)2 * E2 * E2, wv16, E2, E2, KP_K);
    }
    // 2) embedding gather -> f16, and zero the enc16 pad columns
    embed_f16_kernel<<<S, T, 0, stream>>>(x, E_tab, emb16);
    pad_enc16_kernel<<<S, 32, 0, stream>>>(enc16);

    // 3) gi = emb @ Wih.T + bih  for both directions (WMMA)
    //    backward direction consumes emb rows in reverse order
    {
        dim3 grid((G3 + 31) / 32, S / 64);          // 10 x 64
        wmma_gemm_kernel<<<grid, 128, 0, stream>>>(emb16, T, S, 0, wihf16, T, enc_f_bih,
                                                   gi_f, G3, G3, T / 32);
        wmma_gemm_kernel<<<grid, 128, 0, stream>>>(emb16, T, S, 1, wihb16, T, enc_b_bih,
                                                   gi_b, G3, G3, T / 32);
    }
    // 4) sequential GRU scans (persistent blocks, h in LDS)
    gru_recurrent_kernel<<<2, 320, 0, stream>>>(enc_f_Whh, enc_f_bhh, enc_b_Whh, enc_b_bhh,
                                                gi_f, gi_b, enc, enc16);
    // 5) kp/vp = encoded @ Wk.T/Wv.T + bk/bv (WMMA)
    {
        dim3 grid((E2 + 31) / 32, S / 64);          // 7 x 64
        wmma_gemm_kernel<<<grid, 128, 0, stream>>>(enc16, KP_K, S, 0, wk16, KP_K,
                                                   att_in_b + E2, kp, E2, E2, KP_K / 32);
        wmma_gemm_kernel<<<grid, 128, 0, stream>>>(enc16, KP_K, S, 0, wv16, KP_K,
                                                   att_in_b + 2 * E2, vp, E2, E2, KP_K / 32);
    }
    // 6) 50-step attention decoder (single persistent block)
    decoder_kernel<<<1, 1024, 0, stream>>>(enc, kp, vp, mid_W, mid_b,
                                           att_in_W, att_in_b, att_out_W, att_out_b,
                                           dec_Wih, dec_bih, dec_bhh,
                                           last_W, last_b, L_tab, out);
}
